// WKVAttention_68805376082235
// MI455X (gfx1250) — compile-verified, matrix operand order checked by reading, not run on hardware
//
#include <hip/hip_runtime.h>
#include <hip/hip_bf16.h>
#include <stdint.h>

typedef __bf16 bf16;
typedef __attribute__((ext_vector_type(16))) __bf16 v16bf;
typedef __attribute__((ext_vector_type(8)))  __bf16 v8bf;
typedef __attribute__((ext_vector_type(8)))  float  v8f;

#define BATCH   16
#define TLEN    4096
#define CDIM    512
#define NHEADS  8
#define HDIM    64
#define MROWS   (BATCH * TLEN)                 // 65536
#define NELEM   ((size_t)MROWS * CDIM)         // 33,554,432

static __device__ __forceinline__ float sigf(float x) { return 1.0f / (1.0f + expf(-x)); }

// ---------------------------------------------------------------------------
// Kernel 1: bqe channel-mix (4 variants) + plain copy, all emitted as bf16.
// x_star quarter shifts along T: q0:0, q1:+1 (t-1), q2:-1 (t+1), q3:+2 (t-2)
// ---------------------------------------------------------------------------
__global__ __launch_bounds__(256)
void bqe_mix_kernel(const float* __restrict__ x,
                    const float* __restrict__ mur, const float* __restrict__ muk,
                    const float* __restrict__ muv, const float* __restrict__ muw,
                    bf16* __restrict__ xr, bf16* __restrict__ xk,
                    bf16* __restrict__ xv, bf16* __restrict__ xw,
                    bf16* __restrict__ xg)
{
    const size_t idx = (size_t)blockIdx.x * blockDim.x + threadIdx.x;
    const int    c   = (int)(idx & (CDIM - 1));
    const size_t bt  = idx >> 9;
    const int    t   = (int)(bt & (TLEN - 1));
    const size_t b   = bt >> 12;
    const int    q   = c >> 7;

    const float xv0 = x[idx];
    float xs = xv0;
    if (q != 0) {
        const int sh = (q == 1) ? 1 : (q == 2) ? -1 : 2;
        const int ts = (t - sh) & (TLEN - 1);
        xs = x[(b * TLEN + ts) * CDIM + c];
    }
    xr[idx] = (bf16)(xv0 + (1.0f - sigf(mur[c])) * xs);
    xk[idx] = (bf16)(xv0 + (1.0f - sigf(muk[c])) * xs);
    xv[idx] = (bf16)(xv0 + (1.0f - sigf(muv[c])) * xs);
    xw[idx] = (bf16)(xv0 + (1.0f - sigf(muw[c])) * xs);
    xg[idx] = (bf16)xv0;
}

// ---------------------------------------------------------------------------
// WMMA fragment gather from an LDS tile with 32-element row stride.
// ISA 16-bit A/B layout: lanes 0-15 -> row=lane,    K = {0..7, 16..23}
//                        lanes16-31 -> row=lane-16, K = {8..15,24..31}
// ---------------------------------------------------------------------------
static __device__ __forceinline__ v16bf load_frag(const bf16* tile, int rowBase, int lane)
{
    const int half = lane >> 4;
    const int lr   = lane & 15;
    const bf16* rp = tile + (size_t)(rowBase + lr) * 32 + half * 8;
    const v8bf lo  = *(const v8bf*)rp;          // K klo .. klo+7
    const v8bf hi  = *(const v8bf*)(rp + 16);   // K klo+16 .. klo+23
    v16bf f;
#pragma unroll
    for (int i = 0; i < 8; ++i) { f[i] = lo[i]; f[i + 8] = hi[i]; }
    return f;
}

// ---------------------------------------------------------------------------
// Kernel 2: C[M,N] = A[M,K](bf16,row-major) x W[N,K](f32,row-major)^T
// 128x64 tile / workgroup, 8 waves as 4(M) x 2(N), 2x2 subtiles per wave.
// mode 0: store bf16 raw    mode 1: decay w=exp(-exp(base+tanh(v*A)*B)) bf16
// mode 2: sigmoid bf16      mode 3: store f32 raw
// ---------------------------------------------------------------------------
__global__ __launch_bounds__(256)
void gemm_wmma_bf16(const bf16*  __restrict__ A, const float* __restrict__ W,
                    bf16*  __restrict__ outb,  float* __restrict__ outf,
                    const float* __restrict__ dA, const float* __restrict__ dB,
                    const float* __restrict__ dBase, int mode)
{
    __shared__ bf16 Asl[128][32];   // 8 KB
    __shared__ bf16 Bsl[64][32];    // 4 KB, stored [n][k]

    const int tid   = threadIdx.x;
    const int lane  = tid & 31;
    const int wave  = tid >> 5;
    const int waveM = wave >> 1;    // 0..3, 32 rows each
    const int waveN = wave & 1;     // 0..1, 32 cols each
    const int mBase = blockIdx.x * 128;
    const int nBase = blockIdx.y * 64;

    v8f acc[4];
#pragma unroll
    for (int s = 0; s < 4; ++s)
#pragma unroll
        for (int i = 0; i < 8; ++i) acc[s][i] = 0.0f;

    const int arow  = tid >> 1;         // 0..127
    const int ahalf = (tid & 1) * 16;   // 0 or 16

    for (int kc = 0; kc < CDIM / 32; ++kc) {
        const int kBase = kc * 32;
        __syncthreads();
        {   // stage A tile: 128x32 bf16 (each thread: 32 bytes)
            const uint4* src = (const uint4*)(A + (size_t)(mBase + arow) * CDIM + kBase + ahalf);
            uint4* dst = (uint4*)&Asl[arow][ahalf];
            dst[0] = src[0]; dst[1] = src[1];
            if (kc + 1 < CDIM / 32)
                __builtin_prefetch(A + (size_t)(mBase + arow) * CDIM + kBase + 32 + ahalf, 0, 0);
        }
        if (tid < 128) {  // stage B tile: 64x32, f32 -> bf16
            const int brow  = tid >> 1;
            const float* wp = W + (size_t)(nBase + brow) * CDIM + kBase + ahalf;
            bf16* dst = &Bsl[brow][ahalf];
#pragma unroll
            for (int j = 0; j < 16; ++j) dst[j] = (bf16)wp[j];
        }
        __syncthreads();

        const v16bf a0 = load_frag(&Asl[0][0], waveM * 32,      lane);
        const v16bf a1 = load_frag(&Asl[0][0], waveM * 32 + 16, lane);
        const v16bf b0 = load_frag(&Bsl[0][0], waveN * 32,      lane);
        const v16bf b1 = load_frag(&Bsl[0][0], waveN * 32 + 16, lane);

        acc[0] = __builtin_amdgcn_wmma_f32_16x16x32_bf16(false, a0, false, b0, (short)0, acc[0], false, false);
        acc[1] = __builtin_amdgcn_wmma_f32_16x16x32_bf16(false, a0, false, b1, (short)0, acc[1], false, false);
        acc[2] = __builtin_amdgcn_wmma_f32_16x16x32_bf16(false, a1, false, b0, (short)0, acc[2], false, false);
        acc[3] = __builtin_amdgcn_wmma_f32_16x16x32_bf16(false, a1, false, b1, (short)0, acc[3], false, false);
    }

    // Epilogue. C layout: VGPR i -> (M = i + 8*half, N = lane&15) within subtile.
    const int half = lane >> 4;
    const int lr   = lane & 15;
#pragma unroll
    for (int s = 0; s < 4; ++s) {
        const int sm = (s >> 1) * 16;
        const int sn = (s & 1) * 16;
        const int n  = nBase + waveN * 32 + sn + lr;
        float pa = 0.f, pb = 0.f, pc = 0.f;
        if (mode == 1) { pa = dA[n]; pb = dB[n]; pc = dBase[n]; }
#pragma unroll
        for (int i = 0; i < 8; ++i) {
            const int m = mBase + waveM * 32 + sm + half * 8 + i;
            float val = acc[s][i];
            if      (mode == 1) val = expf(-expf(pc + tanhf(val * pa) * pb));
            else if (mode == 2) val = sigf(val);
            if (mode == 3) outf[(size_t)m * CDIM + n] = val;
            else           outb[(size_t)m * CDIM + n] = (bf16)val;
        }
    }
}

// ---------------------------------------------------------------------------
// Kernel 3: WKV linear-attention scan, one workgroup per (b,h,dir).
// State s[d][e] (64x64 f32) in registers: lane owns d = tid>>2, e-slice of 16.
// Faithful to reference: contrib[d] = sum_e s[d][e]*k[e]; bonus = k[d]*u[d]*sum_e v[e];
// s' = w[d]*s + k[d]*v[e]. 64-step chunks staged to LDS; quad shfl_xor reductions.
// ---------------------------------------------------------------------------
__global__ __launch_bounds__(256)
void wkv_scan_kernel(const bf16* __restrict__ r, const bf16* __restrict__ k,
                     const bf16* __restrict__ v, const bf16* __restrict__ w,
                     const float* __restrict__ u,
                     float* __restrict__ out_fwd, float* __restrict__ out_bwd)
{
    __shared__ bf16 rL[64][64], kL[64][64], vL[64][64], wL[64][64];  // 32 KB
    const int tid  = threadIdx.x;
    const int d    = tid >> 2;          // 0..63
    const int esub = tid & 3;
    const int e0   = esub * 16;
    const int bh   = blockIdx.x;        // 0..127
    const int dir  = blockIdx.y;        // 0 fwd, 1 bwd
    const int b    = bh >> 3, h = bh & 7;
    const size_t base = ((size_t)b * TLEN) * CDIM + (size_t)h * HDIM;
    float* outp = dir ? out_bwd : out_fwd;
    const float u_d = u[h * HDIM + d];

    float s[16];
#pragma unroll
    for (int j = 0; j < 16; ++j) s[j] = 0.0f;

    const int lst   = tid >> 2;         // step slot when staging
    const int lpart = (tid & 3) * 16;   // 16-channel quarter

    for (int chunk = 0; chunk < TLEN / 64; ++chunk) {
        __syncthreads();
        {   // cooperative stage of 64 timesteps x 64 channels per tensor
            const int tt = dir ? (TLEN - 1 - (chunk * 64 + lst)) : (chunk * 64 + lst);
            const size_t g = base + (size_t)tt * CDIM + lpart;
            const uint4* rsc = (const uint4*)(r + g);
            const uint4* ksc = (const uint4*)(k + g);
            const uint4* vsc = (const uint4*)(v + g);
            const uint4* wsc = (const uint4*)(w + g);
            uint4* rd_ = (uint4*)&rL[lst][lpart]; rd_[0] = rsc[0]; rd_[1] = rsc[1];
            uint4* kd_ = (uint4*)&kL[lst][lpart]; kd_[0] = ksc[0]; kd_[1] = ksc[1];
            uint4* vd_ = (uint4*)&vL[lst][lpart]; vd_[0] = vsc[0]; vd_[1] = vsc[1];
            uint4* wd_ = (uint4*)&wL[lst][lpart]; wd_[0] = wsc[0]; wd_[1] = wsc[1];
        }
        __syncthreads();
        for (int st = 0; st < 64; ++st) {
            const float kd = (float)kL[st][d];
            const float wd = (float)wL[st][d];
            const float rd = (float)rL[st][d];
            const v8bf kv0 = *(const v8bf*)&kL[st][e0];
            const v8bf kv1 = *(const v8bf*)&kL[st][e0 + 8];
            const v8bf vv0 = *(const v8bf*)&vL[st][e0];
            const v8bf vv1 = *(const v8bf*)&vL[st][e0 + 8];
            float ke[16], ve[16];
#pragma unroll
            for (int j = 0; j < 8; ++j) {
                ke[j] = (float)kv0[j]; ke[j + 8] = (float)kv1[j];
                ve[j] = (float)vv0[j]; ve[j + 8] = (float)vv1[j];
            }
            float contrib = 0.0f, vsum = 0.0f;
#pragma unroll
            for (int j = 0; j < 16; ++j) { contrib += s[j] * ke[j]; vsum += ve[j]; }
            contrib += __shfl_xor(contrib, 1, 32);
            contrib += __shfl_xor(contrib, 2, 32);
            vsum    += __shfl_xor(vsum, 1, 32);
            vsum    += __shfl_xor(vsum, 2, 32);
            const float ot = sigf(rd) * (contrib + kd * u_d * vsum);
            if (esub == 0) {
                const int tt = dir ? (TLEN - 1 - (chunk * 64 + st)) : (chunk * 64 + st);
                outp[base + (size_t)tt * CDIM + d] = ot;
            }
#pragma unroll
            for (int j = 0; j < 16; ++j) s[j] = wd * s[j] + kd * ve[j];
        }
    }
}

// ---------------------------------------------------------------------------
// Kernel 4: GroupNorm over (T, D) per (b,h), fused 0.5*(fwd+bwd), gate-by-g,
// and bf16 re-quantization for the final WMMA GEMM.
// ---------------------------------------------------------------------------
__global__ __launch_bounds__(256)
void gnorm_gate_kernel(const float* __restrict__ of, const float* __restrict__ ob,
                       const bf16*  __restrict__ g,
                       const float* __restrict__ gw, const float* __restrict__ gbias,
                       bf16* __restrict__ act)
{
    __shared__ float rs[256], rq[256];
    const int tid = threadIdx.x;
    const int bh  = blockIdx.x;
    const int b   = bh >> 3, h = bh & 7;
    const size_t base = ((size_t)b * TLEN) * CDIM + (size_t)h * HDIM;
    const int total = TLEN * HDIM;   // 262144

    float sum = 0.0f, sq = 0.0f;
    for (int i = tid; i < total; i += 256) {
        const int t = i >> 6, d = i & 63;
        const size_t idx = base + (size_t)t * CDIM + d;
        const float o = 0.5f * (of[idx] + ob[idx]);
        sum += o; sq += o * o;
    }
    rs[tid] = sum; rq[tid] = sq;
    __syncthreads();
    for (int off = 128; off > 0; off >>= 1) {
        if (tid < off) { rs[tid] += rs[tid + off]; rq[tid] += rq[tid + off]; }
        __syncthreads();
    }
    const float inv_n = 1.0f / (float)total;
    const float mean  = rs[0] * inv_n;
    const float var   = rq[0] * inv_n - mean * mean;
    const float rstd  = rsqrtf(var + 1e-5f);

    for (int i = tid; i < total; i += 256) {
        const int t = i >> 6, d = i & 63;
        const size_t idx = base + (size_t)t * CDIM + d;
        const float o   = 0.5f * (of[idx] + ob[idx]);
        const int   c   = h * HDIM + d;
        const float nrm = (o - mean) * rstd * gw[c] + gbias[c];
        act[idx] = (bf16)(nrm * (float)g[idx]);
    }
}

// ---------------------------------------------------------------------------
extern "C" void kernel_launch(void* const* d_in, const int* in_sizes, int n_in,
                              void* d_out, int out_size, void* d_ws, size_t ws_size,
                              hipStream_t stream)
{
    const float* x     = (const float*)d_in[0];
    const float* mu_r  = (const float*)d_in[1];
    const float* mu_k  = (const float*)d_in[2];
    const float* mu_v  = (const float*)d_in[3];
    const float* mu_w  = (const float*)d_in[4];
    const float* Wr    = (const float*)d_in[5];
    const float* Wk    = (const float*)d_in[6];
    const float* Wv    = (const float*)d_in[7];
    const float* Wg    = (const float*)d_in[8];
    const float* Ww    = (const float*)d_in[9];
    const float* dA    = (const float*)d_in[10];
    const float* dB    = (const float*)d_in[11];
    const float* dBase = (const float*)d_in[12];
    const float* u     = (const float*)d_in[13];
    const float* gn_w  = (const float*)d_in[14];
    const float* gn_b  = (const float*)d_in[15];
    const float* Wout  = (const float*)d_in[16];

    char* ws = (char*)d_ws;
    const size_t SLOT = NELEM * sizeof(bf16);     // 64 MiB
    bf16* xr  = (bf16*)(ws + 0 * SLOT);
    bf16* xk  = (bf16*)(ws + 1 * SLOT);
    bf16* xv  = (bf16*)(ws + 2 * SLOT);
    bf16* xw  = (bf16*)(ws + 3 * SLOT);
    bf16* xg  = (bf16*)(ws + 4 * SLOT);
    bf16* rb  = (bf16*)(ws + 5 * SLOT);
    bf16* kb  = (bf16*)(ws + 6 * SLOT);
    bf16* vb  = (bf16*)(ws + 7 * SLOT);
    bf16* wb  = (bf16*)(ws + 8 * SLOT);
    bf16* gbf = (bf16*)(ws + 9 * SLOT);
    // bqe buffers are dead after the five GEMMs -> alias for f32 scan outputs.
    float* wkv_f = (float*)(ws + 0 * SLOT);       // 128 MiB (slots 0-1)
    float* wkv_b = (float*)(ws + 2 * SLOT);       // 128 MiB (slots 2-3)
    bf16*  act   = (bf16*)(ws + 4 * SLOT);        // slot 4

    // 1) bqe mix -> bf16
    bqe_mix_kernel<<<dim3((unsigned)(NELEM / 256)), 256, 0, stream>>>(
        x, mu_r, mu_k, mu_v, mu_w, xr, xk, xv, xw, xg);

    // 2) five WMMA GEMMs with fused epilogues
    const dim3 gG(MROWS / 128, CDIM / 64);
    gemm_wmma_bf16<<<gG, 256, 0, stream>>>(xr, Wr, rb,  nullptr, nullptr, nullptr, nullptr, 0);
    gemm_wmma_bf16<<<gG, 256, 0, stream>>>(xk, Wk, kb,  nullptr, nullptr, nullptr, nullptr, 0);
    gemm_wmma_bf16<<<gG, 256, 0, stream>>>(xv, Wv, vb,  nullptr, nullptr, nullptr, nullptr, 0);
    gemm_wmma_bf16<<<gG, 256, 0, stream>>>(xw, Ww, wb,  nullptr, dA, dB, dBase,           1);
    gemm_wmma_bf16<<<gG, 256, 0, stream>>>(xg, Wg, gbf, nullptr, nullptr, nullptr, nullptr, 2);

    // 3) bidirectional WKV scan: 256 independent (b,h,dir) workgroups
    wkv_scan_kernel<<<dim3(BATCH * NHEADS, 2), 256, 0, stream>>>(
        rb, kb, vb, wb, u, wkv_f, wkv_b);

    // 4) GroupNorm + gate, emit bf16 activation
    gnorm_gate_kernel<<<dim3(BATCH * NHEADS), 256, 0, stream>>>(
        wkv_f, wkv_b, gbf, gn_w, gn_b, act);

    // 5) final WMMA GEMM -> f32 output
    gemm_wmma_bf16<<<gG, 256, 0, stream>>>(act, Wout, nullptr, (float*)d_out,
                                           nullptr, nullptr, nullptr, 3);
}
